// ChannelFC_10746008175393
// MI455X (gfx1250) — compile-verified, matrix operand order checked by reading, not run on hardware
//
#include <hip/hip_runtime.h>

// Problem constants (from reference): x[B,F,H,W], weight[F,C,C], bias[F,C], C=H*W
#define B_DIM 64
#define F_DIM 64
#define C_DIM 1024
#define NTILE 128   // N columns per workgroup (8 waves x 16)
#define KC    32    // K per stage = one bf16 WMMA K-step

typedef __attribute__((ext_vector_type(4)))  __bf16 v4bf;
typedef __attribute__((ext_vector_type(8)))  __bf16 v8bf;
typedef __attribute__((ext_vector_type(16))) __bf16 v16bf;
typedef __attribute__((ext_vector_type(4)))  float  v4f;
typedef __attribute__((ext_vector_type(8)))  float  v8f;

// Assemble a 16-half WMMA operand from LDS: per-lane K pattern is
// {base .. base+7} in halfs [0..7] and {base+16 .. base+23} in halfs [8..15].
__device__ __forceinline__ v16bf make_op(const __bf16* p) {
  v8bf a = *(const v8bf*)(p);        // K = base..base+7   (16B aligned)
  v8bf b = *(const v8bf*)(p + 16);   // K = base+16..base+23
  return __builtin_shufflevector(a, b, 0, 1, 2, 3, 4, 5, 6, 7,
                                        8, 9, 10, 11, 12, 13, 14, 15);
}

__global__ __launch_bounds__(256)
void channel_fc_wmma(const float* __restrict__ x,
                     const float* __restrict__ w,
                     const float* __restrict__ bias,
                     float* __restrict__ out) {
  // Double-buffered bf16 hi/lo staging: 48 KB total.
  __shared__ __align__(64) __bf16 sAhi[2][B_DIM * KC];
  __shared__ __align__(64) __bf16 sAlo[2][B_DIM * KC];
  __shared__ __align__(64) __bf16 sBhi[2][NTILE * KC];
  __shared__ __align__(64) __bf16 sBlo[2][NTILE * KC];

  const int tid  = threadIdx.x;
  const int f    = blockIdx.y;          // feature index
  const int n0   = blockIdx.x * NTILE;  // N block origin
  const int wave = tid >> 5;            // 0..7 -> owns N tile [n0+16*wave, +16)
  const int lane = tid & 31;

  const size_t xRowStride = (size_t)F_DIM * C_DIM;                 // between b rows
  const float* xBase = x + (size_t)f * C_DIM;                      // A[b][k]
  const float* wBase = w + (size_t)f * C_DIM * C_DIM
                         + (size_t)n0 * C_DIM;                     // B[n][k]

  // Cooperative staging indices: each thread loads float4s from rows
  // (tid/8 + 32*j), columns kk + 4*(tid%8).
  const int row0 = tid >> 3;            // 0..31
  const int col4 = (tid & 7) * 4;       // 0,4,...,28

  v4f aReg[2];
  v4f bReg[4];

  auto stageLoad = [&](int kk) {
    #pragma unroll
    for (int j = 0; j < 2; ++j)
      aReg[j] = *(const v4f*)(xBase + (size_t)(row0 + 32 * j) * xRowStride + kk + col4);
    #pragma unroll
    for (int j = 0; j < 4; ++j)
      bReg[j] = *(const v4f*)(wBase + (size_t)(row0 + 32 * j) * C_DIM + kk + col4);
  };

  auto stageStore = [&](int buf) {
    #pragma unroll
    for (int j = 0; j < 2; ++j) {
      v4bf hi, lo;
      #pragma unroll
      for (int e = 0; e < 4; ++e) {
        float v = aReg[j][e];
        __bf16 h = (__bf16)v;                 // RNE truncation to bf16
        hi[e] = h;
        lo[e] = (__bf16)(v - (float)h);       // residual
      }
      const int off = (row0 + 32 * j) * KC + col4;
      *(v4bf*)(&sAhi[buf][off]) = hi;
      *(v4bf*)(&sAlo[buf][off]) = lo;
    }
    #pragma unroll
    for (int j = 0; j < 4; ++j) {
      v4bf hi, lo;
      #pragma unroll
      for (int e = 0; e < 4; ++e) {
        float v = bReg[j][e];
        __bf16 h = (__bf16)v;
        hi[e] = h;
        lo[e] = (__bf16)(v - (float)h);
      }
      const int off = (row0 + 32 * j) * KC + col4;
      *(v4bf*)(&sBhi[buf][off]) = hi;
      *(v4bf*)(&sBlo[buf][off]) = lo;
    }
  };

  v8f acc[4] = {};                      // 4 M-tiles (M=64) x f32 16x16 accum

  const int rTile = lane & 15;          // row-within-tile (A) / N col (B)
  const int halfK = (lane >> 4) * 8;    // per-lane K base: 0 or 8

  // Prologue: stage chunk 0.
  stageLoad(0);
  stageStore(0);
  __syncthreads();

  const int NSTEP = C_DIM / KC;         // 32
  for (int s = 0; s < NSTEP; ++s) {
    const int cur = s & 1;

    if (s + 1 < NSTEP) stageLoad((s + 1) * KC);
    if (s + 2 < NSTEP)  // hint the k+2 weight lines toward L2
      __builtin_prefetch(wBase + (size_t)row0 * C_DIM + (s + 2) * KC + col4, 0, 0);

    // B operand for this wave's N tile (hi and lo)
    const int bOff = (wave * 16 + rTile) * KC + halfK;
    const v16bf bHi = make_op(&sBhi[cur][bOff]);
    const v16bf bLo = make_op(&sBlo[cur][bOff]);

    #pragma unroll
    for (int m = 0; m < 4; ++m) {
      const int aOff = (m * 16 + rTile) * KC + halfK;
      const v16bf aHi = make_op(&sAhi[cur][aOff]);
      const v16bf aLo = make_op(&sAlo[cur][aOff]);
      // bf16x3 split-precision: hi*hi + hi*lo + lo*hi, f32 accumulate
      acc[m] = __builtin_amdgcn_wmma_f32_16x16x32_bf16(
                   false, aHi, false, bHi, (short)0, acc[m], false, false);
      acc[m] = __builtin_amdgcn_wmma_f32_16x16x32_bf16(
                   false, aHi, false, bLo, (short)0, acc[m], false, false);
      acc[m] = __builtin_amdgcn_wmma_f32_16x16x32_bf16(
                   false, aLo, false, bHi, (short)0, acc[m], false, false);
    }

    if (s + 1 < NSTEP) stageStore((s + 1) & 1);
    __syncthreads();
  }

  // Epilogue: C/D layout -> element r of v8f is row M = r + 8*(lane>=16),
  // lane%16 is the N column. Add bias, write fp32 output.
  const int col = n0 + wave * 16 + rTile;
  const float bv = bias[(size_t)f * C_DIM + col];
  const int hl = (lane >> 4) * 8;
  #pragma unroll
  for (int m = 0; m < 4; ++m) {
    #pragma unroll
    for (int r = 0; r < 8; ++r) {
      const int brow = m * 16 + hl + r;   // batch index b
      out[(size_t)brow * xRowStride + (size_t)f * C_DIM + col] = acc[m][r] + bv;
    }
  }
}

extern "C" void kernel_launch(void* const* d_in, const int* in_sizes, int n_in,
                              void* d_out, int out_size, void* d_ws, size_t ws_size,
                              hipStream_t stream) {
  const float* x    = (const float*)d_in[0];
  const float* wgt  = (const float*)d_in[1];
  const float* bias = (const float*)d_in[2];
  float* out        = (float*)d_out;

  dim3 grid(C_DIM / NTILE, F_DIM);   // (8, 64) = 512 workgroups
  channel_fc_wmma<<<grid, 256, 0, stream>>>(x, wgt, bias, out);

  (void)in_sizes; (void)n_in; (void)out_size; (void)d_ws; (void)ws_size;
}
